// MultiHeadNetWithNestedHistory_34351148434227
// MI455X (gfx1250) — compile-verified, matrix operand order checked by reading, not run on hardware
//
#include <hip/hip_runtime.h>

// ---------------------------------------------------------------- problem dims
#define Bn 512
#define Sn 512
#define Dn 256
#define Qn 32
#define CONDn 64
#define NEG_INF (-1e9f)

typedef __attribute__((ext_vector_type(16))) __bf16 v16bf;
typedef __attribute__((ext_vector_type(8)))  __bf16 v8bf;
typedef __attribute__((ext_vector_type(4)))  __bf16 v4bf;
typedef __attribute__((ext_vector_type(2)))  __bf16 v2bf;
typedef __attribute__((ext_vector_type(8)))  float  v8f;

#define WMMA_BF16(A, B, C) \
    __builtin_amdgcn_wmma_f32_16x16x32_bf16(false, (A), false, (B), (short)0, (C), false, false)

// ----------------------------------------------------------- LDS partitioning
// WB  : weights bf16 [256][256]            131072 B @ 0
// QB  : fused queries bf16 [32][256]        16384 B @ 131072
// KT  : keys tile bf16 [32][256]            16384 B @ 147456
// PT  : projected tile bf16 (16 KB)         16384 B @ 163840
//        pass1: K-tile [32 s][256 d] ; pass2: V-tile transposed [256 d][32 s]
// SC  : scores f32 [32][512]                65536 B @ 180224
// PB  : softmax probs bf16 [32][512]        32768 B @ 245760
// RAW : async-staged keys tile f32 [32][256] 32768 B @ 278528
#define SMEM_BYTES 311296

// -------------------------------------------------- CDNA5 bf16 fragment load
// 16-bit A/B fragment, lane = (half<<4)|m : element e lives at
//   k(e) = 16*(e>>3) + 8*half + (e&7)   -> two contiguous 16B chunks per frag
__device__ __forceinline__ v16bf load_frag_row(const __bf16* row, int kb, int half) {
    v8bf lo = *(const v8bf*)(row + kb + half * 8);
    v8bf hi = *(const v8bf*)(row + kb + 16 + half * 8);
    v16bf r;
#pragma unroll
    for (int i = 0; i < 8; ++i) { r[i] = lo[i]; r[i + 8] = hi[i]; }
    return r;
}

// copy CHUNKS*1024 floats (global or LDS src) -> LDS bf16, 256 threads
template <int CHUNKS>
__device__ __forceinline__ void copy_f32_to_bf16(__bf16* dst, const float* src, int tid) {
#pragma unroll 4
    for (int j = 0; j < CHUNKS; ++j) {
        int flat = j * 1024 + tid * 4;
        float4 v = *(const float4*)(src + flat);
        v4bf o;
        o[0] = (__bf16)v.x; o[1] = (__bf16)v.y; o[2] = (__bf16)v.z; o[3] = (__bf16)v.w;
        *(v4bf*)(dst + flat) = o;
    }
}

// ------------------------------------------ CDNA5 async global->LDS staging
__device__ __forceinline__ void async_load_tile(void* lds_dst, const float* src, int tid) {
    uint32_t base = (uint32_t)(uintptr_t)lds_dst;   // LDS aperture: addr[31:0] = LDS offset
#pragma unroll
    for (int j = 0; j < 8; ++j) {
        int flat = j * 1024 + tid * 4;              // 4 floats (16B) per lane per issue
        asm volatile("global_load_async_to_lds_b128 %0, %1, off"
                     :: "v"(base + (uint32_t)flat * 4u),
                        "v"((unsigned long long)(uintptr_t)(src + flat))
                     : "memory");
    }
}
__device__ __forceinline__ void wait_async() {
    asm volatile("s_wait_asynccnt 0x0" ::: "memory");
}

// --------------------------------------------------- projection result store
__device__ __forceinline__ void store_tile(__bf16* PT, const v8f& acc, int mt, int nt,
                                           int m, int half, bool transpose_out) {
    int col = nt * 16 + m;                          // output-dim index
    if (!transpose_out) {                           // PT[s][d]
#pragma unroll
        for (int r = 0; r < 8; ++r)
            PT[(mt * 16 + r + half * 8) * Dn + col] = (__bf16)acc[r];
    } else {                                        // PT[d][s]; consecutive r adjacent
        __bf16* base = PT + col * 32 + mt * 16 + half * 8;
#pragma unroll
        for (int r = 0; r < 8; r += 2) {
            v2bf p; p[0] = (__bf16)acc[r]; p[1] = (__bf16)acc[r + 1];
            *(v2bf*)(base + r) = p;
        }
    }
}

// PT = KT(32x256) @ W^T, W row-major [e][d] bf16. Per wave: 2x2 output tiles,
// operands loaded once per k-step -> 2 ds_load_b128 per WMMA.
__device__ __forceinline__ void project_tile(const __bf16* WB, const __bf16* KT,
                                             __bf16* PT, int wave, int m, int half,
                                             bool transpose_out) {
    v8f acc0 = {}, acc1 = {}, acc2 = {}, acc3 = {};
    const __bf16* a0 = KT + m * Dn;                       // s-tile 0
    const __bf16* a1 = KT + (16 + m) * Dn;                // s-tile 1
    const __bf16* b0 = WB + (wave * 32 + m) * Dn;         // d-tile 2w
    const __bf16* b1 = WB + (wave * 32 + 16 + m) * Dn;    // d-tile 2w+1
#pragma unroll
    for (int kk = 0; kk < 8; ++kk) {
        v16bf fa0 = load_frag_row(a0, kk * 32, half);
        v16bf fa1 = load_frag_row(a1, kk * 32, half);
        v16bf fb0 = load_frag_row(b0, kk * 32, half);
        v16bf fb1 = load_frag_row(b1, kk * 32, half);
        acc0 = WMMA_BF16(fa0, fb0, acc0);
        acc1 = WMMA_BF16(fa0, fb1, acc1);
        acc2 = WMMA_BF16(fa1, fb0, acc2);
        acc3 = WMMA_BF16(fa1, fb1, acc3);
    }
    store_tile(PT, acc0, 0, wave * 2 + 0, m, half, transpose_out);
    store_tile(PT, acc1, 0, wave * 2 + 1, m, half, transpose_out);
    store_tile(PT, acc2, 1, wave * 2 + 0, m, half, transpose_out);
    store_tile(PT, acc3, 1, wave * 2 + 1, m, half, transpose_out);
}

// ------------------------------------------------------------------- kernel 1
// n_real[b] = sum(mask[b]);  qrow[b][q][d] = (q_flat + ctx@condW^T + b) * D^-.5 * e^-lt
__global__ __launch_bounds__(256) void prep_kernel(
    const unsigned char* __restrict__ mask, const float* __restrict__ context,
    const float* __restrict__ queries, const float* __restrict__ log_temp,
    const float* __restrict__ cond_w, const float* __restrict__ cond_b,
    __bf16* __restrict__ qrow, int* __restrict__ nreal) {
    __shared__ float ctx[CONDn];
    __shared__ int cnt;
    int b = blockIdx.x, tid = threadIdx.x;
    if (tid == 0) cnt = 0;
    if (tid < CONDn) ctx[tid] = context[b * CONDn + tid];
    __syncthreads();
    int local = (int)mask[b * Sn + tid] + (int)mask[b * Sn + 256 + tid];
    atomicAdd(&cnt, local);
    __syncthreads();
    if (tid == 0) nreal[b] = cnt;
    for (int j = 0; j < 32; ++j) {
        int i = j * 256 + tid;                 // 0..8191  (q = i>>8, d = i&255)
        const float* wrow = cond_w + (size_t)i * CONDn;
        float dot = 0.f;
#pragma unroll
        for (int c = 0; c < CONDn; ++c) dot = fmaf(wrow[c], ctx[c], dot);
        float val = queries[i] + dot + cond_b[i];
        float sc  = 0.0625f * __expf(-log_temp[(i >> 8) >> 2]);   // D^-1/2 * inv_t
        qrow[(size_t)b * 8192 + i] = (__bf16)(val * sc);
    }
}

// ------------------------------------------------------------------- kernel 2
__global__ __launch_bounds__(256) void attn_kernel(
    const float* __restrict__ keys, const float* __restrict__ key_w,
    const float* __restrict__ value_w, const __bf16* __restrict__ qrow,
    const int* __restrict__ nreal, float* __restrict__ out) {
    extern __shared__ char smem[];
    __bf16* WB  = (__bf16*)smem;
    __bf16* QB  = (__bf16*)(smem + 131072);
    __bf16* KT  = (__bf16*)(smem + 147456);
    __bf16* PT  = (__bf16*)(smem + 163840);
    float*  SC  = (float*)(smem + 180224);
    __bf16* PB  = (__bf16*)(smem + 245760);
    float*  RAW = (float*)(smem + 278528);

    int b = blockIdx.x, tid = threadIdx.x;
    int wave = tid >> 5, lane = tid & 31;
    int half = lane >> 4, m = lane & 15;
    int nr = nreal[b];
    const float* kbase = keys + (size_t)b * Sn * Dn;

    async_load_tile(RAW, kbase, tid);               // tile 0 in flight ASAP

    {   // fused queries -> LDS (bf16, pre-scaled)
        const uint4* src = (const uint4*)(qrow + (size_t)b * 8192);
        uint4* dst = (uint4*)QB;
#pragma unroll
        for (int j = 0; j < 4; ++j) dst[j * 256 + tid] = src[j * 256 + tid];
    }
    copy_f32_to_bf16<64>(WB, key_w, tid);           // stage Wk

    // ----------------- pass 1: K projection + scores -----------------
    for (int st = 0; st < 16; ++st) {
        wait_async();
        __syncthreads();                            // RAW tile st complete everywhere
        copy_f32_to_bf16<8>(KT, RAW, tid);          // LDS f32 -> LDS bf16
        __syncthreads();
        if (st < 15) async_load_tile(RAW, kbase + (st + 1) * 32 * Dn, tid);
        project_tile(WB, KT, PT, wave, m, half, /*transpose_out=*/false);
        __syncthreads();
        if (wave < 4) {                             // scores: Q(32x256) @ Ktile^T
            int mt = wave >> 1, nt = wave & 1;
            v8f acc = {};
            const __bf16* arow = QB + (mt * 16 + m) * Dn;
            const __bf16* brow = PT + (nt * 16 + m) * Dn;
#pragma unroll
            for (int kk = 0; kk < 8; ++kk) {
                v16bf a  = load_frag_row(arow, kk * 32, half);
                v16bf bb = load_frag_row(brow, kk * 32, half);
                acc = WMMA_BF16(a, bb, acc);
            }
            int s = st * 32 + nt * 16 + m;
            float ga = fmaxf((float)(nr - 1 - s), 0.0f);
            bool valid = s < nr;
            const float slopes[4] = {0.25f, 0.0625f, 0.015625f, 0.00390625f};
#pragma unroll
            for (int r = 0; r < 8; ++r) {           // q&3 == r&3 here
                int q = mt * 16 + r + half * 8;
                SC[q * Sn + s] = valid ? (acc[r] - slopes[r & 3] * ga) : NEG_INF;
            }
        }
        __syncthreads();
    }

    // ----------------- exact full-row softmax -----------------
    for (int rq = 0; rq < 4; ++rq) {
        int q = wave * 4 + rq;
        float* srow = SC + q * Sn;
        float mx = -3.4e38f;
#pragma unroll
        for (int j = 0; j < 16; ++j) mx = fmaxf(mx, srow[lane + j * 32]);
#pragma unroll
        for (int off = 16; off > 0; off >>= 1) mx = fmaxf(mx, __shfl_xor(mx, off, 32));
        float e[16], sum = 0.f;
#pragma unroll
        for (int j = 0; j < 16; ++j) { e[j] = __expf(srow[lane + j * 32] - mx); sum += e[j]; }
#pragma unroll
        for (int off = 16; off > 0; off >>= 1) sum += __shfl_xor(sum, off, 32);
        float rinv = 1.0f / sum;
        __bf16* prow = PB + q * Sn;
#pragma unroll
        for (int j = 0; j < 16; ++j) prow[lane + j * 32] = (__bf16)(e[j] * rinv);
    }
    __syncthreads();
    async_load_tile(RAW, kbase, tid);               // pass-2 tile 0 (RAW idle since p1)
    copy_f32_to_bf16<64>(WB, value_w, tid);         // restage Wv over Wk

    // ----------------- pass 2: V projection + P@V -----------------
    v8f accO[4];
#pragma unroll
    for (int t = 0; t < 4; ++t) accO[t] = (v8f){};
    for (int st = 0; st < 16; ++st) {
        wait_async();
        __syncthreads();
        copy_f32_to_bf16<8>(KT, RAW, tid);
        __syncthreads();
        if (st < 15) async_load_tile(RAW, kbase + (st + 1) * 32 * Dn, tid);
        project_tile(WB, KT, PT, wave, m, half, /*transpose_out=*/true);  // V^T [d][s]
        __syncthreads();
        {   // O[q][d] += P[q][s] * V[s][d] : 2x2 tiles, shared operands
            v16bf pa0 = load_frag_row(PB + m * Sn, st * 32, half);
            v16bf pa1 = load_frag_row(PB + (16 + m) * Sn, st * 32, half);
            v16bf vb0 = load_frag_row(PT + (wave * 32 + m) * 32, 0, half);
            v16bf vb1 = load_frag_row(PT + (wave * 32 + 16 + m) * 32, 0, half);
            accO[0] = WMMA_BF16(pa0, vb0, accO[0]);
            accO[1] = WMMA_BF16(pa0, vb1, accO[1]);
            accO[2] = WMMA_BF16(pa1, vb0, accO[2]);
            accO[3] = WMMA_BF16(pa1, vb1, accO[3]);
        }
        __syncthreads();
    }

    float* ob = out + (size_t)b * 8192;             // [b][q][d] == (B, 8, 4*256) flat
#pragma unroll
    for (int t = 0; t < 4; ++t) {
        int nt = wave * 2 + (t & 1), mt = t >> 1;
#pragma unroll
        for (int r = 0; r < 8; ++r) {
            int q = mt * 16 + r + half * 8;
            ob[q * Dn + nt * 16 + m] = accO[t][r];
        }
    }
}

// ------------------------------------------------------------------- launcher
extern "C" void kernel_launch(void* const* d_in, const int* in_sizes, int n_in,
                              void* d_out, int out_size, void* d_ws, size_t ws_size,
                              hipStream_t stream) {
    const float*         keys    = (const float*)d_in[0];
    const unsigned char* mask    = (const unsigned char*)d_in[1];  // jnp bool -> 1 byte
    const float*         context = (const float*)d_in[2];
    const float*         queries = (const float*)d_in[3];
    const float*         key_w   = (const float*)d_in[4];
    const float*         value_w = (const float*)d_in[5];
    const float*         log_t   = (const float*)d_in[6];
    const float*         cond_w  = (const float*)d_in[7];
    const float*         cond_b  = (const float*)d_in[8];

    __bf16* qrow  = (__bf16*)d_ws;                          // 512*32*256 bf16 = 8 MB
    int*    nreal = (int*)((char*)d_ws + 8388608);          // 512 ints

    prep_kernel<<<Bn, 256, 0, stream>>>(mask, context, queries, log_t, cond_w, cond_b,
                                        qrow, nreal);
    attn_kernel<<<Bn, 256, SMEM_BYTES, stream>>>(keys, key_w, value_w, qrow, nreal,
                                                 (float*)d_out);
}
